// LorentzTransformerEncoder_283467842438
// MI455X (gfx1250) — matrix operand order repaired and checked by reading, not compiled, not run on hardware
//
#include <hip/hip_runtime.h>
#include <hip/hip_fp16.h>
#include <math.h>
#include <stdint.h>

// ---------------------------------------------------------------------------
// Lorentz transformer encoder layer for MI455X (gfx1250, wave32, WMMA).
// All GEMMs use v_wmma_f32_16x16x32_f16. Attention is flash-style fused and
// stages K tiles into LDS with GLOBAL_LOAD_ASYNC_TO_LDS_B128 (ASYNCcnt path).
// Shapes: B=2, N=1024, D=513 (1 time + 512 spatial), H=8, HD=64.
// ---------------------------------------------------------------------------

typedef __attribute__((ext_vector_type(16))) _Float16 v16h;
typedef __attribute__((ext_vector_type(8)))  _Float16 v8h;
typedef __attribute__((ext_vector_type(8)))  float    v8f;

#define WMMA32(a, b, c) \
  __builtin_amdgcn_wmma_f32_16x16x32_f16(false, (a), false, (b), (short)0, (c), false, false)

__device__ __forceinline__ v8f zero8() {
  v8f z;
#pragma unroll
  for (int i = 0; i < 8; ++i) z[i] = 0.0f;
  return z;
}

__device__ __forceinline__ v16h make_a_frag(const _Float16* lo_p, const _Float16* hi_p) {
  v8h lo = *(const v8h*)lo_p;
  v8h hi = *(const v8h*)hi_p;
  v16h a;
#pragma unroll
  for (int i = 0; i < 8; ++i) { a[i] = lo[i]; a[8 + i] = hi[i]; }
  return a;
}

// Async global -> LDS copy of 16 bytes per lane (tracked by ASYNCcnt).
__device__ __forceinline__ void async_load_b128_to_lds(void* lds_dst, const void* gsrc) {
  unsigned lds_addr = (unsigned)(uintptr_t)lds_dst;          // low 32 bits = LDS offset
  unsigned long long ga = (unsigned long long)(uintptr_t)gsrc;
  asm volatile("global_load_async_to_lds_b128 %0, %1, off"
               :: "v"(lds_addr), "v"(ga)
               : "memory");
}

__device__ __forceinline__ void wait_asynccnt0() {
  asm volatile("s_wait_asynccnt 0" ::: "memory");
}

// 256-thread block sum reduction (8 waves of 32).
__device__ __forceinline__ float block_reduce_sum_256(float v, float* sbuf) {
  const int lane = threadIdx.x & 31, w = threadIdx.x >> 5;
#pragma unroll
  for (int o = 16; o > 0; o >>= 1) v += __shfl_xor(v, o, 32);
  if (lane == 0) sbuf[w] = v;
  __syncthreads();
  if (w == 0) {
    float r = (lane < 8) ? sbuf[lane] : 0.0f;
#pragma unroll
    for (int o = 4; o > 0; o >>= 1) r += __shfl_xor(r, o, 32);
    if (lane == 0) sbuf[0] = r;
  }
  __syncthreads();
  float out = sbuf[0];
  __syncthreads();
  return out;
}

// ---------------------------------------------------------------------------
// Generic WMMA GEMM: C[M,N] (f32) = A[M,Kp] (f16) x W[N,Kp] (f16, row-major
// by output col == B^T, i.e. C[m,n] = sum_k A[m,k]*W[n,k]).
// Block: 256 thr = 8 waves; wave tile 16x64; WG tile 128x64.
// ---------------------------------------------------------------------------
__global__ __launch_bounds__(256) void gemm_wmma_f16(
    const _Float16* __restrict__ A, const _Float16* __restrict__ Wt,
    float* __restrict__ C, int M, int N, int Kp) {
  const int lane = threadIdx.x & 31;
  const int wave = threadIdx.x >> 5;
  const int hi = lane >> 4;   // 0/1 half-wave
  const int ln = lane & 15;
  const int rowbase = blockIdx.x * 128 + wave * 16;
  const int colbase = blockIdx.y * 64;

  v8f acc[4];
#pragma unroll
  for (int c = 0; c < 4; ++c) acc[c] = zero8();

  const _Float16* arow = A + (size_t)(rowbase + ln) * Kp;

  for (int k = 0; k < Kp; k += 32) {
    // prefetch the activation stream two tiles ahead (global_prefetch_b8)
    __builtin_prefetch(arow + k + 64, 0, 3);
    v16h a = make_a_frag(arow + k + hi * 16, arow + k + hi * 16 + 8);
#pragma unroll
    for (int c = 0; c < 4; ++c) {
      const _Float16* bp = Wt + (size_t)(colbase + c * 16 + ln) * Kp + k + hi * 16;
      v16h b = *(const v16h*)bp;
      acc[c] = WMMA32(a, b, acc[c]);
    }
  }
#pragma unroll
  for (int c = 0; c < 4; ++c) {
#pragma unroll
    for (int r = 0; r < 8; ++r) {
      const int row = rowbase + r + hi * 8;
      const int col = colbase + c * 16 + ln;
      C[(size_t)row * N + col] = acc[c][r];
    }
  }
}

// ---------------------------------------------------------------------------
// Cast + zero-pad a f32 weight (rows x cols) to f16 (rows x colsPad).
// ---------------------------------------------------------------------------
__global__ void cast_pad_kernel(const float* __restrict__ W, _Float16* __restrict__ Wh,
                                int rows, int cols, int colsPad) {
  const int idx = blockIdx.x * blockDim.x + threadIdx.x;
  if (idx >= rows * colsPad) return;
  const int r = idx / colsPad, c = idx - r * colsPad;
  Wh[idx] = (c < cols) ? (_Float16)W[(size_t)r * cols + c] : (_Float16)0.0f;
}

// ---------------------------------------------------------------------------
// LN1: y1 = add_time(LN(x_spatial)*g+b) -> f16 padded to 544.  One row/block.
// ---------------------------------------------------------------------------
__global__ __launch_bounds__(256) void ln1_kernel(
    const float* __restrict__ x, const float* __restrict__ g,
    const float* __restrict__ b, _Float16* __restrict__ y1h) {
  __shared__ float sbuf[8];
  const int row = blockIdx.x;
  const int t = threadIdx.x;
  const float* xs = x + (size_t)row * 513 + 1;
  const float v0 = xs[t], v1 = xs[t + 256];
  const float mu = block_reduce_sum_256(v0 + v1, sbuf) * (1.0f / 512.0f);
  const float d0 = v0 - mu, d1 = v1 - mu;
  const float var = block_reduce_sum_256(d0 * d0 + d1 * d1, sbuf) * (1.0f / 512.0f);
  const float inv = rsqrtf(var + 1e-5f);
  const float s0 = d0 * inv * g[t] + b[t];
  const float s1 = d1 * inv * g[t + 256] + b[t + 256];
  const float ss = block_reduce_sum_256(s0 * s0 + s1 * s1, sbuf);
  _Float16* out = y1h + (size_t)row * 544;
  if (t == 0) out[0] = (_Float16)sqrtf(1.0f + ss);
  out[1 + t] = (_Float16)s0;
  out[1 + t + 256] = (_Float16)s1;
  if (t < 31) out[513 + t] = (_Float16)0.0f;
}

// ---------------------------------------------------------------------------
// QKV finalize: split heads, add_time for q/k (store t separately), logmap0(v).
// One wave per (b,h,n) row (64 dims), 8 rows per block.
// ---------------------------------------------------------------------------
__global__ __launch_bounds__(256) void qkv_finalize_kernel(
    const float* __restrict__ sq, const float* __restrict__ sk,
    const float* __restrict__ sv, _Float16* __restrict__ qsh,
    float* __restrict__ qt, _Float16* __restrict__ ksh,
    float* __restrict__ kt, _Float16* __restrict__ vth) {
  const int wave = threadIdx.x >> 5, lane = threadIdx.x & 31;
  const int r = blockIdx.x * 8 + wave;        // (b*8+h)*1024 + n
  const int bh = r >> 10, n = r & 1023;
  const int b = bh >> 3, h = bh & 7;
  const size_t src = ((size_t)(b * 1024 + n)) * 512 + h * 64;
  const int d0 = lane * 2;

  float a0 = sq[src + d0], a1 = sq[src + d0 + 1];
  float s = a0 * a0 + a1 * a1;
#pragma unroll
  for (int o = 16; o > 0; o >>= 1) s += __shfl_xor(s, o, 32);
  if (lane == 0) qt[r] = sqrtf(1.0f + s);
  qsh[(size_t)r * 64 + d0] = (_Float16)a0;
  qsh[(size_t)r * 64 + d0 + 1] = (_Float16)a1;

  a0 = sk[src + d0]; a1 = sk[src + d0 + 1];
  s = a0 * a0 + a1 * a1;
#pragma unroll
  for (int o = 16; o > 0; o >>= 1) s += __shfl_xor(s, o, 32);
  if (lane == 0) kt[r] = sqrtf(1.0f + s);
  ksh[(size_t)r * 64 + d0] = (_Float16)a0;
  ksh[(size_t)r * 64 + d0 + 1] = (_Float16)a1;

  a0 = sv[src + d0]; a1 = sv[src + d0 + 1];
  s = a0 * a0 + a1 * a1;
#pragma unroll
  for (int o = 16; o > 0; o >>= 1) s += __shfl_xor(s, o, 32);
  const float vt_time = sqrtf(1.0f + s);
  const float dd = acoshf(fmaxf(vt_time, 1.0f));
  const float sn = fmaxf(sqrtf(s), 1e-8f);
  const float sc = dd / sn;
  vth[(size_t)r * 64 + d0] = (_Float16)(a0 * sc);
  vth[(size_t)r * 64 + d0 + 1] = (_Float16)(a1 * sc);
}

// ---------------------------------------------------------------------------
// Flash hyperbolic attention. Grid (N/128, H, B); 8 waves x 16 query rows.
// K tiles staged via async global->LDS (ASYNCcnt); V transposed via registers.
// ---------------------------------------------------------------------------
__global__ __launch_bounds__(256) void attn_kernel(
    const _Float16* __restrict__ qs, const float* __restrict__ qt,
    const _Float16* __restrict__ ks, const float* __restrict__ kt,
    const _Float16* __restrict__ vt, float* __restrict__ meanTan,
    const float* __restrict__ tempP, const float* __restrict__ betaP,
    const float* __restrict__ tauP, const float* __restrict__ lamP) {
  __shared__ __align__(32) _Float16 Ks[32 * 64];   // [key][feat]
  __shared__ __align__(32) _Float16 Vt[64 * 32];   // [dim][key] (transposed)
  __shared__ float ktl[32];
  __shared__ __align__(32) _Float16 P[8][16 * 32]; // per-wave prob staging

  const int lane = threadIdx.x & 31, wave = threadIdx.x >> 5;
  const int hi = lane >> 4, ln = lane & 15;
  const int bh = blockIdx.z * 8 + blockIdx.y;
  const int qrow0 = blockIdx.x * 128 + wave * 16;
  const size_t baseRow = (size_t)bh * 1024;

  const float beta = log1pf(expf(betaP[0]));
  const float tau = log1pf(expf(tauP[0]));
  const float lam = log1pf(expf(lamP[0]));
  const float invT = 1.0f / tempP[0];
  const float LN2C = 0.6931471805599453f;
  const float PHI0 = log1pf(expf(-0.1f));

  // Q A-fragments (feat chunks 0..31, 32..63)
  v16h aq[2];
  {
    const _Float16* qr = qs + (baseRow + qrow0 + ln) * 64;
#pragma unroll
    for (int c = 0; c < 2; ++c)
      aq[c] = make_a_frag(qr + c * 32 + hi * 16, qr + c * 32 + hi * 16 + 8);
  }
  // per-row precomputed terms (slot r -> row r + hi*8 of the wave tile)
  float qtv[8], Bv[8], nOQ[8], rowmax[8], rowsum[8];
#pragma unroll
  for (int r = 0; r < 8; ++r) {
    const float q = qt[baseRow + qrow0 + r + hi * 8];
    qtv[r] = q;
    const float ct = acoshf(fmaxf(q, 1.001f));
    const float ratio = beta / sinhf(ct);
    Bv[r] = sqrtf(fmaxf(1.0f - ratio * ratio, 0.0f) + 1e-8f);
    nOQ[r] = fmaxf(q * q - 1.0f, 0.0f);
    rowmax[r] = -3.0e38f;
    rowsum[r] = 0.0f;
  }
  v8f acc[4];
#pragma unroll
  for (int c = 0; c < 4; ++c) acc[c] = zero8();

  for (int j = 0; j < 1024; j += 32) {
    // ---- K tile: async copy straight into LDS (256 lanes x 16B = 4KB) ----
    async_load_b128_to_lds(&Ks[threadIdx.x * 8],
                           ks + (baseRow + j) * 64 + threadIdx.x * 8);
    // ---- V tile: load via registers, store transposed [dim][key] ----
    {
      const v8h* vsrc = (const v8h*)(vt + (baseRow + j) * 64);
      v8h vv = vsrc[threadIdx.x];
      const int key = threadIdx.x >> 3, dbase = (threadIdx.x & 7) * 8;
#pragma unroll
      for (int i = 0; i < 8; ++i) Vt[(dbase + i) * 32 + key] = vv[i];
      if (threadIdx.x < 32) ktl[threadIdx.x] = kt[baseRow + j + threadIdx.x];
    }
    wait_asynccnt0();
    __syncthreads();

    // ---- spatial inner products: two 16x16 tiles via WMMA ----
    v8f c0 = zero8(), c1 = zero8();
#pragma unroll
    for (int c = 0; c < 2; ++c) {
      v16h b0 = *(const v16h*)(Ks + ln * 64 + c * 32 + hi * 16);
      v16h b1 = *(const v16h*)(Ks + (16 + ln) * 64 + c * 32 + hi * 16);
      c0 = WMMA32(aq[c], b0, c0);
      c1 = WMMA32(aq[c], b1, c1);
    }
    const float kt0 = ktl[ln], kt1 = ktl[16 + ln];

    // ---- hyperbolic score math + online softmax ----
    float p0v[8], p1v[8];
#pragma unroll
    for (int r = 0; r < 8; ++r) {
      float l0, l1;
#pragma unroll
      for (int e = 0; e < 2; ++e) {
        const float sdot = (e == 0) ? c0[r] : c1[r];
        const float ktx = (e == 0) ? kt0 : kt1;
        const float iqk = sdot - qtv[r] * ktx;
        const float nQK = fmaxf(iqk * iqk - 1.0f, 0.0f);
        const float numer = -(ktx + iqk * qtv[r]);
        const float denom = sqrtf(nQK * nOQ[r] + 2.5e-5f);
        const float Z = fminf(fmaxf(numer / denom, -1.0f), 1.0f);
        const float dist = acoshf(fmaxf(-iqk, 1.001f));
        const float dsoft = 40.0f * tanhf(dist * (1.0f / 40.0f));
        const float sd = dsoft * (1.0f / 15.0f);
        const float Hv = sd + log1pf(__expf(-2.0f * sd)) - LN2C;
        const float Phi = log1pf(__expf(Bv[r] + Z - 0.1f)) - PHI0;
        const float lg = (-lam * Hv - tau * Phi) * invT;
        if (e == 0) l0 = lg; else l1 = lg;
      }
      float tm = fmaxf(l0, l1);
#pragma unroll
      for (int o = 1; o < 16; o <<= 1) tm = fmaxf(tm, __shfl_xor(tm, o, 32));
      const float nm = fmaxf(rowmax[r], tm);
      const float scale = __expf(rowmax[r] - nm);
      rowmax[r] = nm;
      const float e0 = __expf(l0 - nm), e1 = __expf(l1 - nm);
      float es = e0 + e1;
#pragma unroll
      for (int o = 1; o < 16; o <<= 1) es += __shfl_xor(es, o, 32);
      rowsum[r] = rowsum[r] * scale + es;
#pragma unroll
      for (int cd = 0; cd < 4; ++cd) acc[cd][r] *= scale;
      p0v[r] = e0; p1v[r] = e1;
    }

    // ---- restage probabilities (C-layout -> A-layout via per-wave LDS) ----
#pragma unroll
    for (int r = 0; r < 8; ++r) {
      const int rowm = r + hi * 8;
      P[wave][rowm * 32 + ln] = (_Float16)p0v[r];
      P[wave][rowm * 32 + 16 + ln] = (_Float16)p1v[r];
    }
    v16h ap = make_a_frag(&P[wave][ln * 32 + hi * 16], &P[wave][ln * 32 + hi * 16 + 8]);

    // ---- P @ V_tan (4 dim chunks) ----
#pragma unroll
    for (int cd = 0; cd < 4; ++cd) {
      v16h bv = *(const v16h*)(Vt + (cd * 16 + ln) * 32 + hi * 16);
      acc[cd] = WMMA32(ap, bv, acc[cd]);
    }
    __syncthreads();
  }

  // ---- normalize and store tangent-space mean ----
#pragma unroll
  for (int r = 0; r < 8; ++r) {
    const float inv = 1.0f / (rowsum[r] * (1.0f + 1e-8f) + 1e-30f);
    const size_t row = baseRow + qrow0 + r + hi * 8;
#pragma unroll
    for (int cd = 0; cd < 4; ++cd)
      meanTan[row * 64 + cd * 16 + ln] = acc[cd][r] * inv;
  }
}

// ---------------------------------------------------------------------------
// expmap0 + head merge + time recombine -> attn rows (f16, padded 544).
// One (b,n) row per block.
// ---------------------------------------------------------------------------
__global__ __launch_bounds__(256) void combine_kernel(
    const float* __restrict__ meanTan, _Float16* __restrict__ attnh) {
  __shared__ float sums[8];
  const int row = blockIdx.x;           // b*1024 + n
  const int b = row >> 10, n = row & 1023;
  const int lane = threadIdx.x & 31;
  if (threadIdx.x < 8) sums[threadIdx.x] = 0.0f;
  __syncthreads();

  float u[2]; int hh[2];
#pragma unroll
  for (int p = 0; p < 2; ++p) {
    const int e = threadIdx.x + p * 256;
    const int h = e >> 6, d = e & 63;
    hh[p] = h;
    u[p] = meanTan[(((size_t)(b * 8 + h)) * 1024 + n) * 64 + d];
    float v = u[p] * u[p];
#pragma unroll
    for (int o = 16; o > 0; o >>= 1) v += __shfl_xor(v, o, 32);
    if (lane == 0) atomicAdd(&sums[h], v);
  }
  __syncthreads();

  _Float16* out = attnh + (size_t)row * 544;
#pragma unroll
  for (int p = 0; p < 2; ++p) {
    const float nrm = fmaxf(sqrtf(sums[hh[p]]), 1e-8f);
    const float sc = sinhf(nrm) / nrm;
    out[1 + threadIdx.x + p * 256] = (_Float16)(sc * u[p]);
  }
  if (threadIdx.x == 0) {
    float t2 = 0.0f;
#pragma unroll
    for (int h = 0; h < 8; ++h) {
      const float nrm = fmaxf(sqrtf(sums[h]), 1e-8f);
      const float th = coshf(nrm);
      t2 += th * th;
    }
    out[0] = (_Float16)sqrtf(fmaxf(t2 - 7.0f, 1e-8f));
  }
  if (threadIdx.x >= 1 && threadIdx.x < 32) out[513 + threadIdx.x - 1] = (_Float16)0.0f;
}

// ---------------------------------------------------------------------------
// Residual + LN2 -> out_s (f32) and y2 (f16, padded 544).
// ---------------------------------------------------------------------------
__global__ __launch_bounds__(256) void ln2_kernel(
    const float* __restrict__ os, const float* __restrict__ x,
    const float* __restrict__ g, const float* __restrict__ b,
    float* __restrict__ outs, _Float16* __restrict__ y2h) {
  __shared__ float sbuf[8];
  const int row = blockIdx.x;
  const int t = threadIdx.x;
  const float v0 = os[(size_t)row * 512 + t] + x[(size_t)row * 513 + 1 + t];
  const float v1 = os[(size_t)row * 512 + t + 256] + x[(size_t)row * 513 + 1 + t + 256];
  outs[(size_t)row * 512 + t] = v0;
  outs[(size_t)row * 512 + t + 256] = v1;
  const float mu = block_reduce_sum_256(v0 + v1, sbuf) * (1.0f / 512.0f);
  const float d0 = v0 - mu, d1 = v1 - mu;
  const float var = block_reduce_sum_256(d0 * d0 + d1 * d1, sbuf) * (1.0f / 512.0f);
  const float inv = rsqrtf(var + 1e-5f);
  const float s0 = d0 * inv * g[t] + b[t];
  const float s1 = d1 * inv * g[t + 256] + b[t + 256];
  const float ss = block_reduce_sum_256(s0 * s0 + s1 * s1, sbuf);
  _Float16* out = y2h + (size_t)row * 544;
  if (t == 0) out[0] = (_Float16)sqrtf(1.0f + ss);
  out[1 + t] = (_Float16)s0;
  out[1 + t + 256] = (_Float16)s1;
  if (t < 31) out[513 + t] = (_Float16)0.0f;
}

// ---------------------------------------------------------------------------
// GELU (exact) + add_time over 2048 dims -> f16 row padded to 2080.
// ---------------------------------------------------------------------------
__global__ __launch_bounds__(256) void gelu_addtime_kernel(
    const float* __restrict__ h1, _Float16* __restrict__ hh) {
  __shared__ float sbuf[8];
  const int row = blockIdx.x;
  const float* src = h1 + (size_t)row * 2048;
  float g[8];
  float ss = 0.0f;
#pragma unroll
  for (int i = 0; i < 8; ++i) {
    const float z = src[threadIdx.x + i * 256];
    const float gv = 0.5f * z * (1.0f + erff(z * 0.7071067811865475f));
    g[i] = gv;
    ss += gv * gv;
  }
  const float tot = block_reduce_sum_256(ss, sbuf);
  _Float16* out = hh + (size_t)row * 2080;
  if (threadIdx.x == 0) out[0] = (_Float16)sqrtf(1.0f + tot);
#pragma unroll
  for (int i = 0; i < 8; ++i) out[1 + threadIdx.x + i * 256] = (_Float16)g[i];
  if (threadIdx.x < 31) out[2049 + threadIdx.x] = (_Float16)0.0f;
}

// ---------------------------------------------------------------------------
// Final residual + add_time -> d_out (B,N,513) f32.
// ---------------------------------------------------------------------------
__global__ __launch_bounds__(256) void final_kernel(
    const float* __restrict__ f2, const float* __restrict__ outs,
    float* __restrict__ out) {
  __shared__ float sbuf[8];
  const int row = blockIdx.x;
  const int t = threadIdx.x;
  const float v0 = f2[(size_t)row * 512 + t] + outs[(size_t)row * 512 + t];
  const float v1 = f2[(size_t)row * 512 + t + 256] + outs[(size_t)row * 512 + t + 256];
  const float ss = block_reduce_sum_256(v0 * v0 + v1 * v1, sbuf);
  float* o = out + (size_t)row * 513;
  if (t == 0) o[0] = sqrtf(1.0f + ss);
  o[1 + t] = v0;
  o[1 + t + 256] = v1;
}

// ---------------------------------------------------------------------------
extern "C" void kernel_launch(void* const* d_in, const int* in_sizes, int n_in,
                              void* d_out, int out_size, void* d_ws, size_t ws_size,
                              hipStream_t stream) {
  (void)in_sizes; (void)n_in; (void)out_size; (void)ws_size;
  const float* x    = (const float*)d_in[0];
  const float* Wq   = (const float*)d_in[1];
  const float* Wk   = (const float*)d_in[2];
  const float* Wv   = (const float*)d_in[3];
  const float* Wo   = (const float*)d_in[4];
  const float* W1   = (const float*)d_in[5];
  const float* W2   = (const float*)d_in[6];
  const float* ln1g = (const float*)d_in[7];
  const float* ln1b = (const float*)d_in[8];
  const float* ln2g = (const float*)d_in[9];
  const float* ln2b = (const float*)d_in[10];
  const float* temp = (const float*)d_in[11];
  const float* betar= (const float*)d_in[12];
  const float* taur = (const float*)d_in[13];
  const float* lamr = (const float*)d_in[14];

  const int BN = 2048, NT = 16384;  // B*N, B*H*N

  char* ws = (char*)d_ws;
  size_t off = 0;
  auto alloc = [&](size_t bytes) -> void* {
    void* p = ws + off;
    off = (off + bytes + 255) & ~(size_t)255;
    return p;
  };
  _Float16* y1h  = (_Float16*)alloc((size_t)BN * 544 * 2);
  _Float16* Wqh  = (_Float16*)alloc((size_t)512 * 544 * 2);
  _Float16* Wkh  = (_Float16*)alloc((size_t)512 * 544 * 2);
  _Float16* Wvh  = (_Float16*)alloc((size_t)512 * 544 * 2);
  _Float16* Woh  = (_Float16*)alloc((size_t)512 * 544 * 2);
  _Float16* W1h  = (_Float16*)alloc((size_t)2048 * 544 * 2);
  _Float16* W2h  = (_Float16*)alloc((size_t)512 * 2080 * 2);
  float*    sq   = (float*)alloc((size_t)BN * 512 * 4);
  float*    sk   = (float*)alloc((size_t)BN * 512 * 4);
  float*    sv   = (float*)alloc((size_t)BN * 512 * 4);
  _Float16* qsh  = (_Float16*)alloc((size_t)NT * 64 * 2);
  _Float16* ksh  = (_Float16*)alloc((size_t)NT * 64 * 2);
  _Float16* vth  = (_Float16*)alloc((size_t)NT * 64 * 2);
  float*    qt   = (float*)alloc((size_t)NT * 4);
  float*    kt   = (float*)alloc((size_t)NT * 4);
  float*    mtan = (float*)alloc((size_t)NT * 64 * 4);
  _Float16* ath  = (_Float16*)alloc((size_t)BN * 544 * 2);
  float*    osb  = (float*)alloc((size_t)BN * 512 * 4);
  float*    outs = (float*)alloc((size_t)BN * 512 * 4);
  _Float16* y2h  = (_Float16*)alloc((size_t)BN * 544 * 2);
  float*    h1   = (float*)alloc((size_t)BN * 2048 * 4);
  _Float16* hhb  = (_Float16*)alloc((size_t)BN * 2080 * 2);
  float*    f2   = (float*)alloc((size_t)BN * 512 * 4);

  // weights -> f16 padded
  cast_pad_kernel<<<(512 * 544 + 255) / 256, 256, 0, stream>>>(Wq, Wqh, 512, 513, 544);
  cast_pad_kernel<<<(512 * 544 + 255) / 256, 256, 0, stream>>>(Wk, Wkh, 512, 513, 544);
  cast_pad_kernel<<<(512 * 544 + 255) / 256, 256, 0, stream>>>(Wv, Wvh, 512, 513, 544);
  cast_pad_kernel<<<(512 * 544 + 255) / 256, 256, 0, stream>>>(Wo, Woh, 512, 513, 544);
  cast_pad_kernel<<<(2048 * 544 + 255) / 256, 256, 0, stream>>>(W1, W1h, 2048, 513, 544);
  cast_pad_kernel<<<(512 * 2080 + 255) / 256, 256, 0, stream>>>(W2, W2h, 512, 2049, 2080);

  // LN1
  ln1_kernel<<<BN, 256, 0, stream>>>(x, ln1g, ln1b, y1h);

  // QKV projections (WMMA)
  gemm_wmma_f16<<<dim3(16, 8), 256, 0, stream>>>(y1h, Wqh, sq, BN, 512, 544);
  gemm_wmma_f16<<<dim3(16, 8), 256, 0, stream>>>(y1h, Wkh, sk, BN, 512, 544);
  gemm_wmma_f16<<<dim3(16, 8), 256, 0, stream>>>(y1h, Wvh, sv, BN, 512, 544);

  // head split + add_time + logmap0(v)
  qkv_finalize_kernel<<<NT / 8, 256, 0, stream>>>(sq, sk, sv, qsh, qt, ksh, kt, vth);

  // flash hyperbolic attention (WMMA QK^T + WMMA P@V, async K staging)
  attn_kernel<<<dim3(8, 8, 2), 256, 0, stream>>>(qsh, qt, ksh, kt, vth, mtan,
                                                 temp, betar, taur, lamr);

  // expmap0 + head merge
  combine_kernel<<<BN, 256, 0, stream>>>(mtan, ath);

  // output projection (WMMA)
  gemm_wmma_f16<<<dim3(16, 8), 256, 0, stream>>>(ath, Woh, osb, BN, 512, 544);

  // residual + LN2
  ln2_kernel<<<BN, 256, 0, stream>>>(osb, x, ln2g, ln2b, outs, y2h);

  // FFN up (WMMA), GELU+add_time, FFN down (WMMA)
  gemm_wmma_f16<<<dim3(16, 32), 256, 0, stream>>>(y2h, W1h, h1, BN, 2048, 544);
  gelu_addtime_kernel<<<BN, 256, 0, stream>>>(h1, hhb);
  gemm_wmma_f16<<<dim3(16, 8), 256, 0, stream>>>(hhb, W2h, f2, BN, 512, 2080);

  // final residual + add_time
  final_kernel<<<BN, 256, 0, stream>>>(f2, outs, (float*)d_out);
}
